// RobustVeinGNN_55422257987639
// MI455X (gfx1250) — matrix operand order repaired and checked by reading, not compiled, not run on hardware
//
#include <hip/hip_runtime.h>
#include <hip/hip_bf16.h>

typedef __bf16 bf16_t;
typedef __attribute__((ext_vector_type(16))) __bf16 v16bf;
typedef __attribute__((ext_vector_type(8)))  __bf16 v8bf;
typedef __attribute__((ext_vector_type(8)))  float  v8f;

namespace {
constexpr int   NB   = 64;              // graphs
constexpr int   NP   = 1024;            // nodes per graph
constexpr int   DEG  = 8;
constexpr int   NN   = NB * NP;         // 65536 nodes
constexpr int   NE   = NN * DEG;        // 524288 edges
constexpr int   NET  = NE + NN;         // + self loops
constexpr int   HID  = 256;
constexpr int   D3   = 3 * HID;         // 768
constexpr int   NOUT = 128;
constexpr int   KSEL = NP / 2;          // 512 kept per graph
constexpr float EPS  = 1e-5f;
// packed-B geometry: 16 col-tiles x 8 k-steps x 32 lanes x 16 halves
constexpr int   BPACK_ELEMS = 16 * 8 * 32 * 16;   // 65536 per layer
}

// ---------- helpers ----------
__device__ __forceinline__ bf16_t f2bf(float f) {
    unsigned u = __float_as_uint(f);
    unsigned r = (u + 0x7FFFu + ((u >> 16) & 1u)) >> 16;   // RNE
    unsigned short s = (unsigned short)r;
    return __builtin_bit_cast(bf16_t, s);
}

__device__ __forceinline__ void atomicMaxF(float* addr, float v) {
    int* ai = (int*)addr;
    int old = __float_as_int(*addr);
    while (__int_as_float(old) < v) {
        int assumed = old;
        old = atomicCAS(ai, assumed, __float_as_int(v));
        if (old == assumed) break;
    }
}

// ---------- elementwise / prep kernels ----------
__global__ void k_embed(const float* __restrict__ x, const float* __restrict__ W,
                        const float* __restrict__ b, float* __restrict__ h) {
    int idx = blockIdx.x * blockDim.x + threadIdx.x;
    if (idx >= NN * HID) return;
    int n = idx / HID, c = idx % HID;
    h[idx] = x[n * 2 + 0] * W[c] + x[n * 2 + 1] * W[HID + c] + b[c];
}

__global__ void k_tobf16(bf16_t* __restrict__ dst, const float* __restrict__ src,
                         int rows, int cols, int srcStride) {
    int idx = blockIdx.x * blockDim.x + threadIdx.x;
    if (idx >= rows * cols) return;
    int n = idx / cols, c = idx % cols;
    dst[idx] = f2bf(src[n * srcStride + c]);
}

// Pack W (f32 row-major [K=256][N=256]) into per-lane WMMA B-fragment layout (bf16):
// Bp[((ct*8 + ks)*32 + lane)*16 + j] = bf16( W[(ks*32 + (j<8 ? hi*8 + j : 16 + hi*8 + j - 8))*256 + ct*16 + ln] )
__global__ void k_packB(const float* __restrict__ W, bf16_t* __restrict__ Bp) {
    int idx = blockIdx.x * blockDim.x + threadIdx.x;
    if (idx >= BPACK_ELEMS) return;
    int j  = idx & 15;
    int l  = (idx >> 4) & 31;
    int ks = (idx >> 9) & 7;
    int ct = (idx >> 12) & 15;
    int hi = l >> 4, ln = l & 15;
    int kb = (j < 8) ? (hi * 8 + j) : (16 + hi * 8 + (j - 8));
    Bp[idx] = f2bf(W[(ks * 32 + kb) * HID + ct * 16 + ln]);
}

// ---------- WMMA GEMM:  C[M,256] = A(bf16)[M,256] @ W[256,256] (pre-packed) ----------
// block: 256 threads = 8 waves; block tile = 128 rows x 64 cols
// wave: 16 rows x 4 col-tiles (4 accumulators, A fragment reused 4x)
// packed B for the block's 4 col-tiles staged once through LDS (32 KB)
__global__ __launch_bounds__(256) void k_gemm_wmma(const bf16_t* __restrict__ A,
                                                   const bf16_t* __restrict__ Bpack,
                                                   float* __restrict__ C, int M) {
    constexpr int K = HID, NC = HID;
    __shared__ __align__(16) bf16_t Bs[4 * 8 * 32 * 16];   // 32 KB

    // cooperative stage of contiguous packed-B chunk for col-tiles [4*by .. 4*by+3]
    {
        const uint4* srcv = (const uint4*)(Bpack + (size_t)blockIdx.y * (4 * 8 * 32 * 16));
        uint4* dstv = (uint4*)Bs;
#pragma unroll
        for (int i = 0; i < 8; ++i)
            dstv[threadIdx.x + i * 256] = srcv[threadIdx.x + i * 256];
    }
    __syncthreads();

    const int wave = threadIdx.x >> 5;
    const int lane = threadIdx.x & 31;
    const int hi   = lane >> 4;
    const int ln   = lane & 15;
    const int row0 = (blockIdx.x * 8 + wave) * 16;
    if (row0 >= M) return;                       // wave-uniform

    const v16bf* BsV = (const v16bf*)Bs;
    v8f acc[4] = {{}, {}, {}, {}};

    for (int ks = 0; ks < 8; ++ks) {
        const int k0 = ks * 32;
        // A 16x32 bf16 fragment: two contiguous aligned 16B runs per lane
        const bf16_t* ap = A + (size_t)(row0 + ln) * K + k0 + hi * 8;
        v8bf alo = *(const v8bf*)(ap);
        v8bf ahi = *(const v8bf*)(ap + 16);
        v16bf a;
#pragma unroll
        for (int j = 0; j < 8; ++j) { a[j] = alo[j]; a[j + 8] = ahi[j]; }
#pragma unroll
        for (int t = 0; t < 4; ++t) {
            v16bf b = BsV[(t * 8 + ks) * 32 + lane];
            acc[t] = __builtin_amdgcn_wmma_f32_16x16x32_bf16(
                /*neg_a=*/false, a, /*neg_b=*/false, b,
                /*c_mod=*/(short)0, acc[t], /*reuse_a=*/false, /*reuse_b=*/false);
        }
    }

    const int col0 = blockIdx.y * 64;
#pragma unroll
    for (int t = 0; t < 4; ++t)
#pragma unroll
        for (int r = 0; r < 8; ++r)              // D: vgpr r -> row r (+8 hi), col ln
            C[(size_t)(row0 + r + hi * 8) * NC + col0 + t * 16 + ln] = acc[t][r];
}

// ---------- attention score dots:  asr/ads [N,H] ----------
__global__ void k_attn(const float* __restrict__ hW, const float* __restrict__ a_src,
                       const float* __restrict__ a_dst, float* __restrict__ asr,
                       float* __restrict__ ads, int H) {
    int idx = blockIdx.x * blockDim.x + threadIdx.x;
    if (idx >= NN * H) return;
    int n = idx / H, hd = idx % H, C = HID / H;
    const float* hp = hW + n * HID + hd * C;
    float sa = 0.f, sd = 0.f;
    for (int c = 0; c < C; ++c) {
        float v = hp[c];
        sa += v * a_src[hd * C + c];
        sd += v * a_dst[hd * C + c];
    }
    asr[idx] = sa;
    ads[idx] = sd;
}

__global__ void k_init_layer(float* __restrict__ agg, float* __restrict__ m,
                             float* __restrict__ s, int nh) {
    int idx = blockIdx.x * blockDim.x + threadIdx.x;
    if (idx >= NN * HID) return;
    agg[idx] = 0.f;
    if (idx < nh) { m[idx] = -1e30f; s[idx] = 0.f; }
}

// pass 1: e = leaky_relu(asr[src]+ads[dst]); segment max over dst
__global__ void k_edge_max(const int* __restrict__ src, const int* __restrict__ dst,
                           const float* __restrict__ asr, const float* __restrict__ ads,
                           float* __restrict__ ebuf, float* __restrict__ m, int H) {
    int idx = blockIdx.x * blockDim.x + threadIdx.x;
    if (idx >= NET * H) return;
    int e = idx / H, hd = idx % H;
    int s_ = (e < NE) ? src[e] : (e - NE);
    int d_ = (e < NE) ? dst[e] : (e - NE);
    float v = asr[s_ * H + hd] + ads[d_ * H + hd];
    v = (v > 0.f) ? v : 0.2f * v;
    ebuf[idx] = v;
    atomicMaxF(&m[d_ * H + hd], v);
}

// pass 2: ex = exp(e - m[dst]); segment sum over dst
__global__ void k_edge_expsum(const int* __restrict__ src, const int* __restrict__ dst,
                              float* __restrict__ ebuf, const float* __restrict__ m,
                              float* __restrict__ s, int H) {
    int idx = blockIdx.x * blockDim.x + threadIdx.x;
    if (idx >= NET * H) return;
    int e = idx / H, hd = idx % H;
    int d_ = (e < NE) ? dst[e] : (e - NE);
    float ex = expf(ebuf[idx] - m[d_ * H + hd]);
    ebuf[idx] = ex;
    atomicAdd(&s[d_ * H + hd], ex);
    (void)src;
}

// pass 3: agg[dst] += alpha * hW[src]   (256 channels / block-thread)
__global__ __launch_bounds__(256) void k_edge_aggr(const int* __restrict__ src,
                                                   const int* __restrict__ dst,
                                                   const float* __restrict__ ebuf,
                                                   const float* __restrict__ s,
                                                   const float* __restrict__ hW,
                                                   float* __restrict__ agg, int H) {
    constexpr int EPB = 4;
    int base = blockIdx.x * EPB;
    int c = threadIdx.x;
    int cph = HID / H;
    int hd = c / cph;
    for (int i = 0; i < EPB; ++i) {
        int e = base + i;
        if (e >= NET) break;
        int s_ = (e < NE) ? src[e] : (e - NE);
        int d_ = (e < NE) ? dst[e] : (e - NE);
        float alpha = ebuf[e * H + hd] / s[d_ * H + hd];
        atomicAdd(&agg[d_ * HID + c], alpha * hW[s_ * HID + c]);
    }
}

// bias + BN + ReLU -> write into xc column block
__global__ void k_post(const float* __restrict__ agg, const float* __restrict__ bias,
                       const float* __restrict__ g, const float* __restrict__ be,
                       const float* __restrict__ rm, const float* __restrict__ rv,
                       float* __restrict__ xc, int colOff) {
    int idx = blockIdx.x * blockDim.x + threadIdx.x;
    if (idx >= NN * HID) return;
    int n = idx / HID, c = idx % HID;
    float v = agg[idx] + bias[c];
    v = (v - rm[c]) * rsqrtf(rv[c] + EPS) * g[c] + be[c];
    xc[n * D3 + colOff + c] = fmaxf(v, 0.f);
}

// SAG score: srel[n] = xc[n].Wrel ; score[n] = brel + xc[n].Wroot
__global__ void k_sag_dots(const float* __restrict__ xc, const float* __restrict__ Wrel,
                           const float* __restrict__ brel, const float* __restrict__ Wroot,
                           float* __restrict__ srel, float* __restrict__ score) {
    int n = blockIdx.x * blockDim.x + threadIdx.x;
    if (n >= NN) return;
    const float* xp = xc + (size_t)n * D3;
    float a = 0.f, r = 0.f;
    for (int k = 0; k < D3; ++k) {
        float v = xp[k];
        a += v * Wrel[k];
        r += v * Wroot[k];
    }
    srel[n] = a;
    score[n] = r + brel[0];
}

// GraphConv neighbor term (original edges only, no self loops)
__global__ void k_sag_edges(const int* __restrict__ src, const int* __restrict__ dst,
                            const float* __restrict__ srel, float* __restrict__ score) {
    int e = blockIdx.x * blockDim.x + threadIdx.x;
    if (e >= NE) return;
    atomicAdd(&score[dst[e]], srel[src[e]]);
}

// exact top-k(512) per graph via rank counting; stable ties (lower index wins)
__global__ __launch_bounds__(1024) void k_rank_select(const float* __restrict__ score,
                                                      float* __restrict__ wsel) {
    __shared__ float sc[NP];
    int g = blockIdx.x, t = threadIdx.x;
    float me = score[g * NP + t];
    sc[t] = me;
    __syncthreads();
    int rank = 0;
    for (int j = 0; j < NP; ++j) {
        float v = sc[j];
        rank += (v > me) || (v == me && j < t);
    }
    wsel[g * NP + t] = (rank < KSEL) ? tanhf(me) : 0.f;
}

// pooled[g,c] = (1/512) * sum_n wsel[n] * xc[n,c]
__global__ void k_pool(const float* __restrict__ wsel, const float* __restrict__ xc,
                       float* __restrict__ pooled) {
    int g = blockIdx.x;
    int c = blockIdx.y * blockDim.x + threadIdx.x;
    if (c >= D3) return;
    float acc = 0.f;
    const float* xp = xc + (size_t)g * NP * D3 + c;
    const float* wp = wsel + g * NP;
    for (int n = 0; n < NP; ++n) acc += wp[n] * xp[(size_t)n * D3];
    pooled[g * D3 + c] = acc * (1.f / (float)KSEL);
}

__global__ void k_fc(const float* __restrict__ pooled, const float* __restrict__ W,
                     const float* __restrict__ b, float* __restrict__ out) {
    int g = blockIdx.x, o = threadIdx.x;
    float acc = b[o];
    const float* pp = pooled + g * D3;
    for (int k = 0; k < D3; ++k) acc += pp[k] * W[k * NOUT + o];
    out[g * NOUT + o] = acc;
}

// ---------- host driver ----------
extern "C" void kernel_launch(void* const* d_in, const int* in_sizes, int n_in,
                              void* d_out, int out_size, void* d_ws, size_t ws_size,
                              hipStream_t stream) {
    (void)in_sizes; (void)n_in; (void)out_size; (void)ws_size;
    const float* x     = (const float*)d_in[0];
    const int*   ei    = (const int*)d_in[1];
    const int*   src   = ei;
    const int*   dst   = ei + NE;
    const float* emb_W = (const float*)d_in[5];
    const float* emb_b = (const float*)d_in[6];
    const float* Wl[3]  = {(const float*)d_in[7],  (const float*)d_in[15], (const float*)d_in[23]};
    const float* asl[3] = {(const float*)d_in[8],  (const float*)d_in[16], (const float*)d_in[24]};
    const float* adl[3] = {(const float*)d_in[9],  (const float*)d_in[17], (const float*)d_in[25]};
    const float* bl[3]  = {(const float*)d_in[10], (const float*)d_in[18], (const float*)d_in[26]};
    const float* gl[3]  = {(const float*)d_in[11], (const float*)d_in[19], (const float*)d_in[27]};
    const float* bel[3] = {(const float*)d_in[12], (const float*)d_in[20], (const float*)d_in[28]};
    const float* rml[3] = {(const float*)d_in[13], (const float*)d_in[21], (const float*)d_in[29]};
    const float* rvl[3] = {(const float*)d_in[14], (const float*)d_in[22], (const float*)d_in[30]};
    const float* Wrel  = (const float*)d_in[31];
    const float* brel  = (const float*)d_in[32];
    const float* Wroot = (const float*)d_in[33];
    const float* fcW   = (const float*)d_in[34];
    const float* fcb   = (const float*)d_in[35];
    float* out = (float*)d_out;
    const int Hl[3] = {4, 4, 1};

    // workspace carving
    char* wsp = (char*)d_ws;
    size_t off = 0;
    auto carve = [&](size_t bytes) -> char* {
        char* p = wsp + off;
        off = (off + bytes + 255) & ~(size_t)255;
        return p;
    };
    float*  h      = (float*)carve((size_t)NN * HID * 4);
    bf16_t* Abf    = (bf16_t*)carve((size_t)NN * HID * 2);
    bf16_t* Bpack  = (bf16_t*)carve((size_t)3 * BPACK_ELEMS * 2);
    float*  hW     = (float*)carve((size_t)NN * HID * 4);
    float*  xc     = (float*)carve((size_t)NN * D3 * 4);
    float*  asr    = (float*)carve((size_t)NN * 4 * 4);
    float*  ads    = (float*)carve((size_t)NN * 4 * 4);
    float*  mbuf   = (float*)carve((size_t)NN * 4 * 4);
    float*  sbuf   = (float*)carve((size_t)NN * 4 * 4);
    float*  ebuf   = (float*)carve((size_t)NET * 4 * 4);
    float*  agg    = (float*)carve((size_t)NN * HID * 4);
    float*  srel   = (float*)carve((size_t)NN * 4);
    float*  score  = (float*)carve((size_t)NN * 4);
    float*  wsel   = (float*)carve((size_t)NN * 4);
    float*  pooled = (float*)carve((size_t)NB * D3 * 4);

    const int TH = 256;

    // node embedding
    k_embed<<<(NN * HID) / TH, TH, 0, stream>>>(x, emb_W, emb_b, h);

    // pack weights into WMMA B-fragment layout (bf16), once per call
    for (int l = 0; l < 3; ++l)
        k_packB<<<BPACK_ELEMS / TH, TH, 0, stream>>>(Wl[l], Bpack + (size_t)l * BPACK_ELEMS);

    // three GAT layers
    for (int l = 0; l < 3; ++l) {
        const int H = Hl[l];
        const float* Xin = (l == 0) ? h : (xc + (l - 1) * HID);
        const int stride = (l == 0) ? HID : D3;

        k_tobf16<<<(NN * HID) / TH, TH, 0, stream>>>(Abf, Xin, NN, HID, stride);
        k_gemm_wmma<<<dim3(NN / 128, 4), TH, 0, stream>>>(
            Abf, Bpack + (size_t)l * BPACK_ELEMS, hW, NN);
        k_attn<<<(NN * H + TH - 1) / TH, TH, 0, stream>>>(hW, asl[l], adl[l], asr, ads, H);
        k_init_layer<<<(NN * HID) / TH, TH, 0, stream>>>(agg, mbuf, sbuf, NN * H);
        k_edge_max<<<(NET * H + TH - 1) / TH, TH, 0, stream>>>(src, dst, asr, ads, ebuf, mbuf, H);
        k_edge_expsum<<<(NET * H + TH - 1) / TH, TH, 0, stream>>>(src, dst, ebuf, mbuf, sbuf, H);
        k_edge_aggr<<<(NET + 3) / 4, TH, 0, stream>>>(src, dst, ebuf, sbuf, hW, agg, H);
        k_post<<<(NN * HID) / TH, TH, 0, stream>>>(agg, bl[l], gl[l], bel[l], rml[l], rvl[l],
                                                   xc, l * HID);
    }

    // SAGPooling score (linear -> scalar-per-node trick)
    k_sag_dots<<<(NN + TH - 1) / TH, TH, 0, stream>>>(xc, Wrel, brel, Wroot, srel, score);
    k_sag_edges<<<(NE + TH - 1) / TH, TH, 0, stream>>>(src, dst, srel, score);
    k_rank_select<<<NB, NP, 0, stream>>>(score, wsel);
    k_pool<<<dim3(NB, D3 / TH), TH, 0, stream>>>(wsel, xc, pooled);
    k_fc<<<NB, NOUT, 0, stream>>>(pooled, fcW, fcb, out);
}
